// NodeEdgeGNN_39642548142790
// MI455X (gfx1250) — matrix operand order repaired and checked by reading, not compiled, not run on hardware
//
#include <hip/hip_runtime.h>

#define NNODES 50000
#define NEDGES 500000
#define NGRAPHS 64
#define BNEPS 1e-5f

typedef __attribute__((ext_vector_type(16))) __bf16 v16bf;
typedef __attribute__((ext_vector_type(8)))  float  v8f;

// ---------------- workspace layout (bytes) ----------------
constexpr long long OFF_H     = 0;                                  // [N x 64] f32
constexpr long long OFF_E1    = OFF_H     + (long long)NNODES*64*4; // [E x 64] f32
constexpr long long OFF_TE    = OFF_E1    + (long long)NEDGES*64*4; // [E x 64] f32 scratch
constexpr long long OFF_TN    = OFF_TE    + (long long)NEDGES*64*4; // [N x 64] f32 scratch
constexpr long long OFF_AGG   = OFF_TN    + (long long)NNODES*64*4; // [N x 64] f32
constexpr long long OFF_STATS = OFF_AGG   + (long long)NNODES*64*4; // [128] f32 (sum, sumsq)
constexpr long long OFF_MV    = OFF_STATS + 128*4;                  // [128] f32 (scale, shift)
constexpr long long OFF_GM    = OFF_MV    + 128*4;                  // [64 x 128] f32 pooled
constexpr long long OFF_CNT   = OFF_GM    + 64*128*4;               // [128] f32 counts
constexpr long long OFF_WBF   = OFF_CNT   + 128*4;                  // bf16 weights, swizzled

__device__ __forceinline__ v8f wmma_bf16(v16bf a, v16bf b, v8f c) {
  return __builtin_amdgcn_wmma_f32_16x16x32_bf16(false, a, false, b, (short)0, c, false, false);
}

// One 16x32 A fragment (this lane's share): two contiguous 8-float runs at
// rowBase+kaBase and rowBase+kaBase+16, converted to bf16.
__device__ __forceinline__ v16bf fragA(const float* rowBase, int kaBase) {
  float4 x0 = *(const float4*)(rowBase + kaBase);
  float4 x1 = *(const float4*)(rowBase + kaBase + 4);
  float4 x2 = *(const float4*)(rowBase + kaBase + 16);
  float4 x3 = *(const float4*)(rowBase + kaBase + 20);
  v16bf a;
  a[0]=(__bf16)x0.x;  a[1]=(__bf16)x0.y;  a[2]=(__bf16)x0.z;  a[3]=(__bf16)x0.w;
  a[4]=(__bf16)x1.x;  a[5]=(__bf16)x1.y;  a[6]=(__bf16)x1.z;  a[7]=(__bf16)x1.w;
  a[8]=(__bf16)x2.x;  a[9]=(__bf16)x2.y;  a[10]=(__bf16)x2.z; a[11]=(__bf16)x2.w;
  a[12]=(__bf16)x3.x; a[13]=(__bf16)x3.y; a[14]=(__bf16)x3.z; a[15]=(__bf16)x3.w;
  return a;
}

// Shared epilogue: bias add, store pre-activations, per-column sum/sumsq into LDS.
__device__ __forceinline__ void store_stats(const v8f* acc, int m0, int lane,
                                            const float* __restrict__ bias,
                                            float* __restrict__ P,
                                            float* sSum, float* sSq) {
  int lrow = lane & 15;
  int mrow = m0 + ((lane < 16) ? 0 : 8);
#pragma unroll
  for (int t = 0; t < 4; ++t) {
    int n = t * 16 + lrow;
    float bv = bias[n];
    float s = 0.f, q = 0.f;
#pragma unroll
    for (int r = 0; r < 8; ++r) {
      float v = acc[t][r] + bv;
      P[(mrow + r) * 64 + n] = v;
      s += v; q += v * v;
    }
    atomicAdd(&sSum[n], s);
    atomicAdd(&sSq[n], q);
  }
}

// ---------------- weight conversion fp32 -> bf16, swizzled to WMMA B layout --------
// dst element idx = (((ks*4 + t)*32 + lane)*16 + i)  holds  W[k][n] with
//   k = ks*32 + (lane<16 ? 0 : 8) + (i<8 ? i : i+8),  n = t*16 + (lane&15)
struct CvtEnt { const float* src; int rows; int padRows; int dstOff; };
struct CvtTab { CvtEnt e[16]; };

__global__ __launch_bounds__(256) void k_cvt(CvtTab tab, __bf16* dst) {
  CvtEnt en = tab.e[blockIdx.x];
  int total = en.padRows * 64;
  for (int idx = threadIdx.x; idx < total; idx += blockDim.x) {
    int i    = idx & 15;
    int lane = (idx >> 4) & 31;
    int t    = (idx >> 9) & 3;
    int ks   = idx >> 11;
    int k = ks * 32 + ((lane < 16) ? 0 : 8) + ((i < 8) ? i : i + 8);
    int n = t * 16 + (lane & 15);
    float v = (k < en.rows) ? en.src[k * 64 + n] : 0.f;
    dst[en.dstOff + idx] = (__bf16)v;
  }
}

__global__ __launch_bounds__(256) void k_zero(float* p, int n) {
  int i = blockIdx.x * blockDim.x + threadIdx.x;
  if (i < n) p[i] = 0.f;
}

// ---------------- GEMM1 variants: P = A @ W1(bf16) + b, with column stats ----------
// Edge message MLP layer 1: A row = [H[dst] | H[src] | E1[row]], K = 192.
__global__ __launch_bounds__(256) void k_gemm_edge(const float* __restrict__ H,
                                                   const float* __restrict__ E1,
                                                   const int* __restrict__ dst,
                                                   const int* __restrict__ src,
                                                   const __bf16* __restrict__ Wb,
                                                   const float* __restrict__ bias,
                                                   float* __restrict__ P,
                                                   float* __restrict__ stats, int M) {
  __shared__ float sSum[64], sSq[64];
  int lane = threadIdx.x & 31, wave = threadIdx.x >> 5;
  if (threadIdx.x < 64) { sSum[threadIdx.x] = 0.f; sSq[threadIdx.x] = 0.f; }
  __syncthreads();
  int m0 = (blockIdx.x * 8 + wave) * 16;
  if (m0 < M) {
    int mA = m0 + (lane & 15);
    int kaBase = (lane < 16) ? 0 : 8;
    int rd = dst[mA], rs = src[mA];
    const float* seg0 = H  + rd * 64;
    const float* seg1 = H  + rs * 64;
    const float* seg2 = E1 + mA * 64;
    const v16bf* Wt = (const v16bf*)Wb;
    v8f acc[4] = {};
    const float* segs[3] = { seg0, seg1, seg2 };
#pragma unroll
    for (int ks = 0; ks < 6; ++ks) {
      v16bf a = fragA(segs[ks >> 1] + (ks & 1) * 32, kaBase);
#pragma unroll
      for (int t = 0; t < 4; ++t)
        acc[t] = wmma_bf16(a, Wt[(ks * 4 + t) * 32 + lane], acc[t]);
    }
    store_stats(acc, m0, lane, bias, P, sSum, sSq);
  }
  __syncthreads();
  if (threadIdx.x < 64) {
    atomicAdd(&stats[threadIdx.x],      sSum[threadIdx.x]);
    atomicAdd(&stats[64 + threadIdx.x], sSq[threadIdx.x]);
  }
}

// Node update MLP layer 1: A row = [H[row] | AGG[row]], K = 128.
__global__ __launch_bounds__(256) void k_gemm_node(const float* __restrict__ H,
                                                   const float* __restrict__ AGG,
                                                   const __bf16* __restrict__ Wb,
                                                   const float* __restrict__ bias,
                                                   float* __restrict__ P,
                                                   float* __restrict__ stats, int M) {
  __shared__ float sSum[64], sSq[64];
  int lane = threadIdx.x & 31, wave = threadIdx.x >> 5;
  if (threadIdx.x < 64) { sSum[threadIdx.x] = 0.f; sSq[threadIdx.x] = 0.f; }
  __syncthreads();
  int m0 = (blockIdx.x * 8 + wave) * 16;
  if (m0 < M) {
    int mA = m0 + (lane & 15);
    int kaBase = (lane < 16) ? 0 : 8;
    const float* segs[2] = { H + mA * 64, AGG + mA * 64 };
    const v16bf* Wt = (const v16bf*)Wb;
    v8f acc[4] = {};
#pragma unroll
    for (int ks = 0; ks < 4; ++ks) {
      v16bf a = fragA(segs[ks >> 1] + (ks & 1) * 32, kaBase);
#pragma unroll
      for (int t = 0; t < 4; ++t)
        acc[t] = wmma_bf16(a, Wt[(ks * 4 + t) * 32 + lane], acc[t]);
    }
    store_stats(acc, m0, lane, bias, P, sSum, sSq);
  }
  __syncthreads();
  if (threadIdx.x < 64) {
    atomicAdd(&stats[threadIdx.x],      sSum[threadIdx.x]);
    atomicAdd(&stats[64 + threadIdx.x], sSq[threadIdx.x]);
  }
}

// Embedder MLP layer 1: dense A [M x W0], K padded to 32. For W0==16 the upper
// half-fragment (k in [16,32)) is zero for BOTH lane halves at compile time.
template <int W0>
__global__ __launch_bounds__(256) void k_gemm_dense(const float* __restrict__ X,
                                                    const __bf16* __restrict__ Wb,
                                                    const float* __restrict__ bias,
                                                    float* __restrict__ P,
                                                    float* __restrict__ stats, int M) {
  __shared__ float sSum[64], sSq[64];
  int lane = threadIdx.x & 31, wave = threadIdx.x >> 5;
  if (threadIdx.x < 64) { sSum[threadIdx.x] = 0.f; sSq[threadIdx.x] = 0.f; }
  __syncthreads();
  int m0 = (blockIdx.x * 8 + wave) * 16;
  if (m0 < M) {
    int mA = m0 + (lane & 15);
    int kaBase = (lane < 16) ? 0 : 8;
    const float* rb = X + mA * W0;
    float4 x0 = *(const float4*)(rb + kaBase);
    float4 x1 = *(const float4*)(rb + kaBase + 4);
    v16bf a;
    a[0]=(__bf16)x0.x; a[1]=(__bf16)x0.y; a[2]=(__bf16)x0.z; a[3]=(__bf16)x0.w;
    a[4]=(__bf16)x1.x; a[5]=(__bf16)x1.y; a[6]=(__bf16)x1.z; a[7]=(__bf16)x1.w;
    if (W0 == 32) {
      float4 x2 = *(const float4*)(rb + kaBase + 16);
      float4 x3 = *(const float4*)(rb + kaBase + 20);
      a[8]=(__bf16)x2.x;  a[9]=(__bf16)x2.y;  a[10]=(__bf16)x2.z; a[11]=(__bf16)x2.w;
      a[12]=(__bf16)x3.x; a[13]=(__bf16)x3.y; a[14]=(__bf16)x3.z; a[15]=(__bf16)x3.w;
    } else {
#pragma unroll
      for (int i = 8; i < 16; ++i) a[i] = (__bf16)0.f;
    }
    const v16bf* Wt = (const v16bf*)Wb;
    v8f acc[4] = {};
#pragma unroll
    for (int t = 0; t < 4; ++t)
      acc[t] = wmma_bf16(a, Wt[t * 32 + lane], acc[t]);
    store_stats(acc, m0, lane, bias, P, sSum, sSq);
  }
  __syncthreads();
  if (threadIdx.x < 64) {
    atomicAdd(&stats[threadIdx.x],      sSum[threadIdx.x]);
    atomicAdd(&stats[64 + threadIdx.x], sSq[threadIdx.x]);
  }
}

// ---------------- finalize BN: scale/shift per column ----------------
__global__ void k_finstats(const float* __restrict__ stats, const float* __restrict__ g,
                           const float* __restrict__ be, float* __restrict__ mv, float Mf) {
  int c = threadIdx.x;
  float mean = stats[c] / Mf;
  float var  = stats[64 + c] / Mf - mean * mean;
  float s = g[c] * rsqrtf(var + BNEPS);
  mv[c]      = s;
  mv[64 + c] = be[c] - mean * s;
}

// ---------------- GEMM2: OUT = [relu]( relu(P*s+t) @ Wf + bf ) [+ resid] -----------
__global__ __launch_bounds__(256) void k_bn_gemm(const float* __restrict__ P,
                                                 const float* __restrict__ mv,
                                                 const __bf16* __restrict__ Wfb,
                                                 const float* __restrict__ bfv,
                                                 const float* __restrict__ resid,
                                                 float* __restrict__ OUT,
                                                 int M, int finalRelu) {
  __shared__ float sMV[128];
  if (threadIdx.x < 128) sMV[threadIdx.x] = mv[threadIdx.x];
  __syncthreads();

  int lane = threadIdx.x & 31, wave = threadIdx.x >> 5;
  int m0   = (blockIdx.x * 8 + wave) * 16;
  if (m0 >= M) return;
  int lrow   = lane & 15;
  int mA     = m0 + lrow;
  int kaBase = (lane < 16) ? 0 : 8;

  const v16bf* Wt = (const v16bf*)Wfb;
  const float* prow = P + mA * 64;
  v8f acc[4] = {};
#pragma unroll
  for (int ks = 0; ks < 2; ++ks) {
    int kb = ks * 32 + kaBase;
    float4 a0 = *(const float4*)(prow + kb);
    float4 a1 = *(const float4*)(prow + kb + 4);
    float4 a2 = *(const float4*)(prow + kb + 16);
    float4 a3 = *(const float4*)(prow + kb + 20);
    float r0[8] = {a0.x,a0.y,a0.z,a0.w,a1.x,a1.y,a1.z,a1.w};
    float r1[8] = {a2.x,a2.y,a2.z,a2.w,a3.x,a3.y,a3.z,a3.w};
    v16bf a;
#pragma unroll
    for (int i = 0; i < 8; ++i) {
      int k0i = kb + i, k1i = kb + 16 + i;
      a[i]     = (__bf16)fmaxf(r0[i] * sMV[k0i] + sMV[64 + k0i], 0.f);
      a[8 + i] = (__bf16)fmaxf(r1[i] * sMV[k1i] + sMV[64 + k1i], 0.f);
    }
#pragma unroll
    for (int t = 0; t < 4; ++t)
      acc[t] = wmma_bf16(a, Wt[(ks * 4 + t) * 32 + lane], acc[t]);
  }
  int mrow = m0 + ((lane < 16) ? 0 : 8);
#pragma unroll
  for (int t = 0; t < 4; ++t) {
    int n = t * 16 + lrow;
    float bv = bfv[n];
#pragma unroll
    for (int r = 0; r < 8; ++r) {
      float v = acc[t][r] + bv;
      if (finalRelu) v = fmaxf(v, 0.f);
      if (resid) v += resid[(mrow + r) * 64 + n];
      OUT[(mrow + r) * 64 + n] = v;
    }
  }
}

// ---------------- scatter-sum e -> nodes (by dst) ----------------
__global__ __launch_bounds__(256) void k_scatter(const float* __restrict__ e,
                                                 const int* __restrict__ dst,
                                                 float* __restrict__ agg) {
  int t = blockIdx.x * blockDim.x + threadIdx.x;
  if (t < NEDGES * 64) {
    int ed = t >> 6, c = t & 63;
    atomicAdd(&agg[dst[ed] * 64 + c], e[t]);
  }
}

// ---------------- graph pooling ----------------
__global__ __launch_bounds__(256) void k_pool_nodes(const float* __restrict__ H,
                                                    const int* __restrict__ batch,
                                                    float* __restrict__ GM, float* __restrict__ CNT) {
  int t = blockIdx.x * blockDim.x + threadIdx.x;
  if (t < NNODES * 64) {
    int i = t >> 6, c = t & 63;
    int g = batch[i];
    atomicAdd(&GM[g * 128 + c], H[t]);
    if (c == 0) atomicAdd(&CNT[g], 1.f);
  }
}
__global__ __launch_bounds__(256) void k_pool_edges(const float* __restrict__ E1,
                                                    const int* __restrict__ src,
                                                    const int* __restrict__ batch,
                                                    float* __restrict__ GM, float* __restrict__ CNT) {
  int t = blockIdx.x * blockDim.x + threadIdx.x;
  if (t < NEDGES * 64) {
    int ed = t >> 6, c = t & 63;
    int g = batch[src[ed]];
    atomicAdd(&GM[g * 128 + 64 + c], E1[t]);
    if (c == 0) atomicAdd(&CNT[64 + g], 1.f);
  }
}

// ---------------- final readout MLP (64x128 -> 64 -> 32 -> 1, BN+ReLU hidden) -------
__global__ __launch_bounds__(256) void k_final(const float* __restrict__ GM, const float* __restrict__ CNT,
                                               const float* W1, const float* b1, const float* be1, const float* g1,
                                               const float* W2, const float* b2, const float* be2, const float* g2,
                                               const float* Wf, const float* bfv,
                                               float* __restrict__ out) {
  __shared__ float A[64 * 128];
  __shared__ float B[64 * 64];
  __shared__ float C[64 * 32];
  __shared__ float red[128];
  int tid = threadIdx.x;
  for (int i = tid; i < 64 * 128; i += 256) {
    int g = i >> 7, c = i & 127;
    float cnt = (c < 64) ? CNT[g] : CNT[64 + g];
    A[i] = GM[i] / fmaxf(cnt, 1.f);
  }
  __syncthreads();
  for (int i = tid; i < 64 * 64; i += 256) {
    int g = i >> 6, j = i & 63;
    float s = b1[j];
    for (int k = 0; k < 128; ++k) s += A[g * 128 + k] * W1[k * 64 + j];
    B[i] = s;
  }
  __syncthreads();
  if (tid < 64) {
    float s = 0.f, q = 0.f;
    for (int g = 0; g < 64; ++g) { float v = B[g * 64 + tid]; s += v; q += v * v; }
    float mean = s / 64.f, var = q / 64.f - mean * mean;
    float sc = g1[tid] * rsqrtf(var + BNEPS);
    red[tid] = sc; red[64 + tid] = be1[tid] - mean * sc;
  }
  __syncthreads();
  for (int i = tid; i < 64 * 64; i += 256) { int j = i & 63; B[i] = fmaxf(B[i] * red[j] + red[64 + j], 0.f); }
  __syncthreads();
  for (int i = tid; i < 64 * 32; i += 256) {
    int g = i >> 5, j = i & 31;
    float s = b2[j];
    for (int k = 0; k < 64; ++k) s += B[g * 64 + k] * W2[k * 32 + j];
    C[i] = s;
  }
  __syncthreads();
  if (tid < 32) {
    float s = 0.f, q = 0.f;
    for (int g = 0; g < 64; ++g) { float v = C[g * 32 + tid]; s += v; q += v * v; }
    float mean = s / 64.f, var = q / 64.f - mean * mean;
    float sc = g2[tid] * rsqrtf(var + BNEPS);
    red[tid] = sc; red[64 + tid] = be2[tid] - mean * sc;
  }
  __syncthreads();
  for (int i = tid; i < 64 * 32; i += 256) { int j = i & 31; C[i] = fmaxf(C[i] * red[j] + red[64 + j], 0.f); }
  __syncthreads();
  if (tid < 64) {
    float s = bfv[0];
    for (int k = 0; k < 32; ++k) s += C[tid * 32 + k] * Wf[k];
    out[tid] = s;
  }
}

// ---------------- host orchestration ----------------
extern "C" void kernel_launch(void* const* d_in, const int* in_sizes, int n_in,
                              void* d_out, int out_size, void* d_ws, size_t ws_size,
                              hipStream_t stream) {
  (void)in_sizes; (void)n_in; (void)out_size; (void)ws_size;
  auto F = [&](int i) { return (const float*)d_in[i]; };
  auto I = [&](int i) { return (const int*)d_in[i]; };
  char* ws = (char*)d_ws;
  float*  H     = (float*)(ws + OFF_H);
  float*  E1    = (float*)(ws + OFF_E1);
  float*  TE    = (float*)(ws + OFF_TE);
  float*  TN    = (float*)(ws + OFF_TN);
  float*  AGG   = (float*)(ws + OFF_AGG);
  float*  STATS = (float*)(ws + OFF_STATS);
  float*  MV    = (float*)(ws + OFF_MV);
  float*  GM    = (float*)(ws + OFF_GM);
  float*  CNT   = (float*)(ws + OFF_CNT);
  __bf16* WBF   = (__bf16*)(ws + OFF_WBF);

  // inputs: 0=x 1=edge_attr 2=edge_index(2xE) 3=batch, then params pytree leaves:
  // emb_edges: 4=Wf 5=bf 6=W 7=b 8=be 9=g ; emb_nodes: 10..15 ;
  // final: 16=Wf 17=bf 18=h0W 19=h0b 20=h0be 21=h0g 22=h1W 23=h1b 24=h1be 25=h1g ;
  // proc[l]: base=26+12l: edge{Wf,bf,W,b,be,g} node{Wf,bf,W,b,be,g}
  const int* srcI  = I(2);
  const int* dstI  = I(2) + NEDGES;
  const int* batch = I(3);

  CvtTab tab;
  tab.e[0] = { F(12), 32, 32, 0 };     // emb_nodes W1
  tab.e[1] = { F(10), 64, 64, 2048 };  // emb_nodes Wf
  tab.e[2] = { F(6),  16, 32, 6144 };  // emb_edges W1 (K padded 16->32)
  tab.e[3] = { F(4),  64, 64, 8192 };  // emb_edges Wf
  for (int l = 0; l < 3; ++l) {
    int b = 26 + 12 * l, off = 12288 + 28672 * l;
    tab.e[4 + 4 * l + 0] = { F(b + 2), 192, 192, off };          // edge W1
    tab.e[4 + 4 * l + 1] = { F(b + 0),  64,  64, off + 12288 };  // edge Wf
    tab.e[4 + 4 * l + 2] = { F(b + 8), 128, 128, off + 16384 };  // node W1
    tab.e[4 + 4 * l + 3] = { F(b + 6),  64,  64, off + 24576 };  // node Wf
  }
  k_cvt<<<dim3(16), dim3(256), 0, stream>>>(tab, WBF);

  const int NBLK = (NNODES + 127) / 128;
  const int EBLK = (NEDGES + 127) / 128;

  auto bnStage = [&](int M, const float* gg, const float* be,
                     const __bf16* Wfb, const float* bfv,
                     float* P, const float* resid, float* OUTp, int fRelu) {
    k_finstats<<<dim3(1), dim3(64), 0, stream>>>(STATS, gg, be, MV, (float)M);
    int blocks = (M + 127) / 128;
    k_bn_gemm<<<dim3(blocks), dim3(256), 0, stream>>>(P, MV, Wfb, bfv, resid, OUTp, M, fRelu);
  };

  // node embedder: x[N,32] -> H
  k_zero<<<dim3(1), dim3(256), 0, stream>>>(STATS, 128);
  k_gemm_dense<32><<<dim3(NBLK), dim3(256), 0, stream>>>(F(0), WBF + 0, F(13), TN, STATS, NNODES);
  bnStage(NNODES, F(15), F(14), WBF + 2048, F(11), TN, nullptr, H, 0);

  // edge embedder: edge_attr[E,16] (K padded to 32) -> E1
  k_zero<<<dim3(1), dim3(256), 0, stream>>>(STATS, 128);
  k_gemm_dense<16><<<dim3(EBLK), dim3(256), 0, stream>>>(F(1), WBF + 6144, F(7), TE, STATS, NEDGES);
  bnStage(NEDGES, F(9), F(8), WBF + 8192, F(5), TE, nullptr, E1, 0);

  for (int l = 0; l < 3; ++l) {
    int b = 26 + 12 * l, off = 12288 + 28672 * l;
    // edge update: e = relu(MLP([h[dst] | h[src] | e])) + e
    k_zero<<<dim3(1), dim3(256), 0, stream>>>(STATS, 128);
    k_gemm_edge<<<dim3(EBLK), dim3(256), 0, stream>>>(H, E1, dstI, srcI,
        WBF + off, F(b + 3), TE, STATS, NEDGES);
    bnStage(NEDGES, F(b + 5), F(b + 4), WBF + off + 12288, F(b + 1), TE, E1, E1, 1);

    // aggregate messages
    k_zero<<<dim3((NNODES * 64 + 255) / 256), dim3(256), 0, stream>>>(AGG, NNODES * 64);
    k_scatter<<<dim3((NEDGES * 64 + 255) / 256), dim3(256), 0, stream>>>(E1, dstI, AGG);

    // node update: h = relu(MLP([h | agg])) + h
    k_zero<<<dim3(1), dim3(256), 0, stream>>>(STATS, 128);
    k_gemm_node<<<dim3(NBLK), dim3(256), 0, stream>>>(H, AGG,
        WBF + off + 16384, F(b + 9), TN, STATS, NNODES);
    bnStage(NNODES, F(b + 11), F(b + 10), WBF + off + 24576, F(b + 7), TN, H, H, 1);
  }

  // pooling (GM and CNT are contiguous in ws)
  k_zero<<<dim3((64 * 128 + 128 + 255) / 256), dim3(256), 0, stream>>>(GM, 64 * 128 + 128);
  k_pool_nodes<<<dim3((NNODES * 64 + 255) / 256), dim3(256), 0, stream>>>(H, batch, GM, CNT);
  k_pool_edges<<<dim3((NEDGES * 64 + 255) / 256), dim3(256), 0, stream>>>(E1, srcI, batch, GM, CNT);

  k_final<<<dim3(1), dim3(256), 0, stream>>>(GM, CNT,
      F(18), F(19), F(20), F(21), F(22), F(23), F(24), F(25), F(16), F(17),
      (float*)d_out);
}
